// MetaLayerGNN_71768903516633
// MI455X (gfx1250) — compile-verified, matrix operand order checked by reading, not run on hardware
//
#include <hip/hip_runtime.h>

typedef _Float16 half_t;
typedef half_t v16h __attribute__((ext_vector_type(16)));
typedef float  v8f  __attribute__((ext_vector_type(8)));

#define BN_EPS 1e-5f

// ------------------------------------------------------------------
// utility kernels
// ------------------------------------------------------------------
__global__ void zero_f32(float* p, int n) {
    int i = blockIdx.x * blockDim.x + threadIdx.x;
    if (i < n) p[i] = 0.0f;
}

__global__ void count_kernel(const int* idx, float* cnt, int n) {
    int i = blockIdx.x * blockDim.x + threadIdx.x;
    if (i < n) atomicAdd(&cnt[idx[i]], 1.0f);
}

// vals: rows x 64 (stride 64). dst[idx[r]*64+c] += vals[r*64+c]
__global__ void scatter_add_kernel(const float* vals, const int* idx,
                                   float* dst, int rows) {
    int i = blockIdx.x * blockDim.x + threadIdx.x;
    if (i < rows * 64) {
        int r = i >> 6;
        int c = i & 63;
        atomicAdd(&dst[idx[r] * 64 + c], vals[i]);
    }
}

__global__ void divide_kernel(float* acc, const float* cnt, int rows) {
    int i = blockIdx.x * blockDim.x + threadIdx.x;
    if (i < rows * 64) acc[i] /= fmaxf(cnt[i >> 6], 1.0f);
}

// one block per column; computes mean and rsqrt(var+eps)
__global__ void bn_stats_kernel(const float* t, int rows, int cols, float* stats) {
    __shared__ float ssum[256];
    __shared__ float ssq[256];
    int col = blockIdx.x;
    float s = 0.0f, q = 0.0f;
    for (int r = threadIdx.x; r < rows; r += blockDim.x) {
        float v = t[(long)r * cols + col];
        s += v;
        q += v * v;
    }
    ssum[threadIdx.x] = s;
    ssq[threadIdx.x]  = q;
    __syncthreads();
    for (int st = 128; st > 0; st >>= 1) {
        if ((int)threadIdx.x < st) {
            ssum[threadIdx.x] += ssum[threadIdx.x + st];
            ssq[threadIdx.x]  += ssq[threadIdx.x + st];
        }
        __syncthreads();
    }
    if (threadIdx.x == 0) {
        float m = ssum[0] / (float)rows;
        float v = ssq[0] / (float)rows - m * m;
        stats[col]        = m;
        stats[cols + col] = rsqrtf(v + BN_EPS);
    }
}

__global__ void bn_apply_kernel(const float* in, float* out, const float* stats,
                                const float* gamma, const float* beta,
                                int rows, int cols, int ostride) {
    long i = (long)blockIdx.x * blockDim.x + threadIdx.x;
    if (i < (long)rows * cols) {
        int r = (int)(i / cols);
        int c = (int)(i % cols);
        float v = in[i];
        out[(long)r * ostride + c] =
            gamma[c] * (v - stats[c]) * stats[cols + c] + beta[c];
    }
}

// ------------------------------------------------------------------
// WMMA GEMM kernel: out[Mrows x 64] = relu(gather(Mrows x din) @ W + bias)
//   mode 0 (edge):   gather = [x[src], x[dst], e_row, u[batch[src]]]
//   mode 1 (node):   gather = [x_row, agg_row, u[batch]]
//   mode 2 (global): gather = [gmean_row, u_row]
// Per wave: one M-tile of 16 rows, N = 64 (4 WMMA column tiles),
// K swept in chunks of 32 with v_wmma_f32_16x16x32_f16.
// Weights are staged in LDS in WMMA B-fragment order:
//   frag id i = (c*4 + nt)*32 + lane, 16 contiguous halves per fragment,
//   element h = W[(c*32 + (lane>>4)*8 + h + (h>=8?8:0)) * 64 + nt*16 + (lane&15)]
// so each B fragment is two ds_load_b128 per lane.
// ------------------------------------------------------------------
__global__ __launch_bounds__(256) void gemm_wmma_kernel(
    int mode, int Mrows, int din, int ew, int uw,
    const int* src, const int* dst, const int* batch,
    const float* xbuf, const float* ebuf, const float* ubuf,
    const float* agg, const float* gmean,
    const float* W, const float* bias, float* out)
{
    // max din(256) x 64 halves = 32 KB, fragment-order, 32B aligned
    __shared__ __attribute__((aligned(32))) half_t wlds[256 * 64];

    int tid = threadIdx.x;

    // fill LDS with f32->f16 converted, fragment-swizzled weights
    int nfrag = (din >> 5) * 4 * 32;   // (c, nt, lane) triples
    for (int i = tid; i < nfrag; i += blockDim.x) {
        int laneI = i & 31;
        int nt    = (i >> 5) & 3;
        int c     = i >> 7;
        int n     = nt * 16 + (laneI & 15);
        int kg    = laneI >> 4;
        half_t* dp = &wlds[i * 16];
#pragma unroll
        for (int h = 0; h < 16; h++) {
            int k = c * 32 + kg * 8 + h + ((h >= 8) ? 8 : 0);
            dp[h] = (half_t)W[k * 64 + n];
        }
    }
    __syncthreads();

    int wave = tid >> 5;
    int lane = tid & 31;
    int tile = blockIdx.x * 8 + wave;
    int row0 = tile * 16;
    if (row0 >= Mrows) return;

    int m  = lane & 15;   // M row within tile (for A gather)
    int kg = lane >> 4;   // K-half selector
    int n  = lane & 15;   // N column within tile (for B / C / D)

    // per-lane gather segments for row row0+m
    int r = row0 + m;
    if (r >= Mrows) r = Mrows - 1;
    const float *p0, *p1, *p2, *p3;
    int w0, w1, w2;
    if (mode == 0) {
        int s = src[r], d = dst[r];
        p0 = xbuf + (long)s * 64; w0 = 64;
        p1 = xbuf + (long)d * 64; w1 = 64;
        p2 = ebuf + (long)r * 64; w2 = ew;
        p3 = ubuf + (long)batch[s] * 64;
    } else if (mode == 1) {
        p0 = xbuf + (long)r * 64; w0 = 64;
        p1 = agg  + (long)r * 64; w1 = 64;
        p2 = ubuf + (long)batch[r] * 64; w2 = uw;
        p3 = p0;
    } else {
        p0 = gmean + (long)r * 64; w0 = 64;
        p1 = ubuf  + (long)r * 64; w1 = uw;
        p2 = p0; w2 = 1 << 30;
        p3 = p0;
    }

    v8f acc0, acc1, acc2, acc3;
    float b0 = bias[n], b1 = bias[16 + n], b2 = bias[32 + n], b3 = bias[48 + n];
#pragma unroll
    for (int j = 0; j < 8; j++) {
        acc0[j] = b0; acc1[j] = b1; acc2[j] = b2; acc3[j] = b3;
    }

    const v16h* wfrag = (const v16h*)wlds;
    int nchunks = din >> 5;
    for (int c = 0; c < nchunks; c++) {
        int kbase = c * 32 + kg * 8;

        // A fragment (16x32 f16), built from gathered f32 features;
        // each 8-run [kbase, kbase+8) and [kbase+16, kbase+24) stays inside
        // one segment (all boundaries are multiples of 32), so the compiler
        // vectorizes these into global_load_b128 + v_cvt_pk_f16_f32.
        v16h a;
#pragma unroll
        for (int h = 0; h < 16; h++) {
            int f = kbase + h + ((h >= 8) ? 8 : 0);
            float v;
            if (f < w0) v = p0[f];
            else {
                int f1 = f - w0;
                if (f1 < w1) v = p1[f1];
                else {
                    int f2 = f1 - w1;
                    if (f2 < w2) v = p2[f2];
                    else         v = p3[f2 - w2];
                }
            }
            a[h] = (half_t)v;
        }

        // B fragments from LDS: two ds_load_b128 each
        int fbase = (c * 4) * 32 + lane;
        v16h bb0 = wfrag[fbase];
        v16h bb1 = wfrag[fbase + 32];
        v16h bb2 = wfrag[fbase + 64];
        v16h bb3 = wfrag[fbase + 96];
        acc0 = __builtin_amdgcn_wmma_f32_16x16x32_f16(false, a, false, bb0,
                                                      (short)0, acc0, false, false);
        acc1 = __builtin_amdgcn_wmma_f32_16x16x32_f16(false, a, false, bb1,
                                                      (short)0, acc1, false, false);
        acc2 = __builtin_amdgcn_wmma_f32_16x16x32_f16(false, a, false, bb2,
                                                      (short)0, acc2, false, false);
        acc3 = __builtin_amdgcn_wmma_f32_16x16x32_f16(false, a, false, bb3,
                                                      (short)0, acc3, false, false);
    }

    // store with fused ReLU; C/D layout: lanes 0-15 -> M=j, lanes 16-31 -> M=8+j
    int mbase = kg * 8;
#pragma unroll
    for (int j = 0; j < 8; j++) {
        int r2 = row0 + mbase + j;
        if (r2 < Mrows) {
            float* o = out + (long)r2 * 64 + n;
            o[0]  = fmaxf(acc0[j], 0.0f);
            o[16] = fmaxf(acc1[j], 0.0f);
            o[32] = fmaxf(acc2[j], 0.0f);
            o[48] = fmaxf(acc3[j], 0.0f);
        }
    }
}

// ------------------------------------------------------------------
// host orchestration
// ------------------------------------------------------------------
extern "C" void kernel_launch(void* const* d_in, const int* in_sizes, int n_in,
                              void* d_out, int out_size, void* d_ws, size_t ws_size,
                              hipStream_t stream) {
    constexpr int N = 30000, E = 300000, B = 64;

    const float* x      = (const float*)d_in[0];
    const int*   ei     = (const int*)  d_in[1];
    const float* e      = (const float*)d_in[2];
    const float* u      = (const float*)d_in[3];
    const int*   batch  = (const int*)  d_in[4];
    const float* g_node = (const float*)d_in[5];
    const float* b_node = (const float*)d_in[6];
    const float* g_edge = (const float*)d_in[7];
    const float* b_edge = (const float*)d_in[8];
    const float* g_glob = (const float*)d_in[9];
    const float* b_glob = (const float*)d_in[10];

    const int* srcp = ei;
    const int* dstp = ei + E;

    // weights: l=0..2, per l: We,be,Wn,bn,Wg,bg starting at index 11
    const float* We[3]; const float* be[3];
    const float* Wn[3]; const float* bn[3];
    const float* Wg[3]; const float* bg[3];
    for (int l = 0; l < 3; l++) {
        We[l] = (const float*)d_in[11 + 6 * l + 0];
        be[l] = (const float*)d_in[11 + 6 * l + 1];
        Wn[l] = (const float*)d_in[11 + 6 * l + 2];
        bn[l] = (const float*)d_in[11 + 6 * l + 3];
        Wg[l] = (const float*)d_in[11 + 6 * l + 4];
        bg[l] = (const float*)d_in[11 + 6 * l + 5];
    }

    const int de[3] = {192, 256, 256};
    const int dn[3] = {160, 192, 192};
    const int dg[3] = { 96, 128, 128};

    // workspace layout (floats)
    float* ws = (float*)d_ws;
    long off = 0;
    float* xbuf  = ws + off; off += (long)N * 64;
    float* ebuf  = ws + off; off += (long)E * 64;
    float* ubuf  = ws + off; off += (long)B * 64;
    float* agg   = ws + off; off += (long)N * 64;
    float* cnt   = ws + off; off += N;
    float* gmean = ws + off; off += (long)B * 64;
    float* gcnt  = ws + off; off += B;
    float* stats = ws + off; off += 256;

    const int T = 256;
    auto blocks = [](long n, int t) { return (int)((n + t - 1) / t); };

    // degree / graph-size counts (fixed across layers)
    zero_f32<<<blocks(N, T), T, 0, stream>>>(cnt, N);
    zero_f32<<<blocks(B, T), T, 0, stream>>>(gcnt, B);
    count_kernel<<<blocks(E, T), T, 0, stream>>>(dstp, cnt, E);
    count_kernel<<<blocks(N, T), T, 0, stream>>>(batch, gcnt, N);

    // input BatchNorm (training-mode batch stats)
    bn_stats_kernel<<<64, 256, 0, stream>>>(x, N, 64, stats);
    bn_apply_kernel<<<blocks((long)N * 64, T), T, 0, stream>>>(
        x, xbuf, stats, g_node, b_node, N, 64, 64);
    bn_stats_kernel<<<32, 256, 0, stream>>>(e, E, 32, stats);
    bn_apply_kernel<<<blocks((long)E * 32, T), T, 0, stream>>>(
        e, ebuf, stats, g_edge, b_edge, E, 32, 64);
    bn_stats_kernel<<<32, 256, 0, stream>>>(u, B, 32, stats);
    bn_apply_kernel<<<blocks((long)B * 32, T), T, 0, stream>>>(
        u, ubuf, stats, g_glob, b_glob, B, 32, 64);

    for (int l = 0; l < 3; l++) {
        int ew = (l == 0) ? 32 : 64;
        int uw = (l == 0) ? 32 : 64;

        // edge model (in-place update of ebuf -> width 64)
        {
            int tiles = (E + 15) / 16;
            int blk = (tiles + 7) / 8;
            gemm_wmma_kernel<<<blk, 256, 0, stream>>>(
                0, E, de[l], ew, uw, srcp, dstp, batch,
                xbuf, ebuf, ubuf, agg, gmean, We[l], be[l], ebuf);
        }
        // segment mean of new edge features at dst
        zero_f32<<<blocks((long)N * 64, T), T, 0, stream>>>(agg, N * 64);
        scatter_add_kernel<<<blocks((long)E * 64, T), T, 0, stream>>>(ebuf, dstp, agg, E);
        divide_kernel<<<blocks((long)N * 64, T), T, 0, stream>>>(agg, cnt, N);

        // node model (in-place update of xbuf)
        {
            int tiles = (N + 15) / 16;
            int blk = (tiles + 7) / 8;
            gemm_wmma_kernel<<<blk, 256, 0, stream>>>(
                1, N, dn[l], ew, uw, srcp, dstp, batch,
                xbuf, ebuf, ubuf, agg, gmean, Wn[l], bn[l], xbuf);
        }
        // per-graph mean of new node features
        zero_f32<<<blocks((long)B * 64, T), T, 0, stream>>>(gmean, B * 64);
        scatter_add_kernel<<<blocks((long)N * 64, T), T, 0, stream>>>(xbuf, batch, gmean, N);
        divide_kernel<<<blocks((long)B * 64, T), T, 0, stream>>>(gmean, gcnt, B);

        // global model (in-place update of ubuf -> width 64)
        gemm_wmma_kernel<<<1, 256, 0, stream>>>(
            2, B, dg[l], ew, uw, srcp, dstp, batch,
            xbuf, ebuf, ubuf, agg, gmean, Wg[l], bg[l], ubuf);
    }

    // outputs: (x, e, u) concatenated flat, f32
    float* out = (float*)d_out;
    hipMemcpyAsync(out, xbuf, (size_t)N * 64 * sizeof(float),
                   hipMemcpyDeviceToDevice, stream);
    hipMemcpyAsync(out + (long)N * 64, ebuf, (size_t)E * 64 * sizeof(float),
                   hipMemcpyDeviceToDevice, stream);
    hipMemcpyAsync(out + (long)N * 64 + (long)E * 64, ubuf,
                   (size_t)B * 64 * sizeof(float),
                   hipMemcpyDeviceToDevice, stream);
}
